// MultiHeadAttention_79448305041611
// MI455X (gfx1250) — matrix operand order repaired and checked by reading, not compiled, not run on hardware
//
#include <hip/hip_runtime.h>
#include <hip/hip_bf16.h>

#define D_MODEL 1024
#define N_HEADS 16
#define D_K     64
#define SEQ     2048
#define BATCH   2
#define TOKENS  (BATCH * SEQ)

typedef __bf16 bf16_t;
typedef __attribute__((ext_vector_type(16))) __bf16 v16bf;
typedef __attribute__((ext_vector_type(8)))  float  v8f;

union FragU { v16bf v; uint4 u[2]; };
union Pack8 { bf16_t h[8]; uint4 u; };

// ---- WMMA wrapper: D = A(16x32 bf16) * B(32x16 bf16) + C(16x16 f32) ----
__device__ __forceinline__ v8f wmma_bf16(v16bf a, v16bf b, v8f c) {
  return __builtin_amdgcn_wmma_f32_16x16x32_bf16(
      /*neg_a=*/false, a, /*neg_b=*/false, b,
      /*c_mod=*/(short)0, c, /*reuse_a=*/false, /*reuse_b=*/false);
}

// ---- A-operand fragment (16x32, 16-bit). Per ISA table:
// lane half g holds K in runs [8g, 8g+8) (VGPR0-3) and [16+8g, 16+8g+8) (VGPR4-7).
__device__ __forceinline__ v16bf load_a_bf16(const bf16_t* row, int g) {
  FragU f;
  f.u[0] = *(const uint4*)(row + g * 8);
  f.u[1] = *(const uint4*)(row + 16 + g * 8);
  return f.v;
}

// ---- B-operand fragment (32x16, 16-bit): lane = column,
// lane half g holds K in runs [8g, 8g+8) and [16+8g, 16+8g+8) (same as A).
__device__ __forceinline__ v16bf load_b_bf16(const bf16_t* col, int g) {
  FragU f;
  f.u[0] = *(const uint4*)(col + g * 8);
  f.u[1] = *(const uint4*)(col + 16 + g * 8);
  return f.v;
}

// =====================================================================
// Kernel 0: fp32 -> bf16 bulk convert (X and the four weight matrices).
// =====================================================================
__global__ __launch_bounds__(256) void cvt_bf16(const float* __restrict__ src,
                                                bf16_t* __restrict__ dst,
                                                int n) {
  const int i = (blockIdx.x * blockDim.x + threadIdx.x) * 8;
  if (i >= n) return;
  const float4* p = (const float4*)(src + i);
  float4 a = p[0], b = p[1];
  Pack8 o;
  o.h[0]=(bf16_t)a.x; o.h[1]=(bf16_t)a.y; o.h[2]=(bf16_t)a.z; o.h[3]=(bf16_t)a.w;
  o.h[4]=(bf16_t)b.x; o.h[5]=(bf16_t)b.y; o.h[6]=(bf16_t)b.z; o.h[7]=(bf16_t)b.w;
  *(uint4*)(dst + i) = o.u;
}

// =====================================================================
// Kernel 1: y = x @ W.T + b (bf16 operands, f32 accum).
// Block tile 128x128 (8 waves as 4M x 2N), wave tile 32x64 (2x4 WMMA
// tiles): 8 WMMAs per 12 b128 loads.
// mode 0: Q -> [B,H,S,D] scaled by 1/sqrt(D_K)
// mode 1: K -> [B,H,S,D]
// mode 2: V -> [B,H,D,S]  (transposed so PV B-fragments are contiguous)
// =====================================================================
__global__ __launch_bounds__(256) void qkv_gemm(
    const bf16_t* __restrict__ Xb, const bf16_t* __restrict__ Wb,
    const float* __restrict__ bias, bf16_t* __restrict__ out,
    int mode, float scale) {
  const int lane = threadIdx.x & 31;
  const int wave = threadIdx.x >> 5;
  const int g    = lane >> 4;
  const int ln   = lane & 15;
  const int wm   = wave >> 1;          // 4 waves along M
  const int wn   = wave & 1;           // 2 waves along N
  const int m0   = blockIdx.x * 128 + wm * 32;
  const int n0   = blockIdx.y * 128 + wn * 64;

  const bf16_t* xr0 = Xb + (size_t)(m0 + ln) * D_MODEL;
  const bf16_t* xr1 = Xb + (size_t)(m0 + 16 + ln) * D_MODEL;
  const bf16_t* wr[4] = {
    Wb + (size_t)(n0 + ln) * D_MODEL,
    Wb + (size_t)(n0 + 16 + ln) * D_MODEL,
    Wb + (size_t)(n0 + 32 + ln) * D_MODEL,
    Wb + (size_t)(n0 + 48 + ln) * D_MODEL,
  };

  v8f acc[2][4] = {};

  #pragma unroll 2
  for (int kk = 0; kk < D_MODEL; kk += 32) {
    __builtin_prefetch(xr0 + kk + 256, 0, 3);     // WGP-scope stream prefetch
    __builtin_prefetch(wr[0] + kk + 256, 0, 3);
    const v16bf a0 = load_a_bf16(xr0 + kk, g);
    const v16bf a1 = load_a_bf16(xr1 + kk, g);
    #pragma unroll
    for (int ti = 0; ti < 4; ++ti) {
      const v16bf b = load_b_bf16(wr[ti] + kk, g);
      acc[0][ti] = wmma_bf16(a0, b, acc[0][ti]);
      acc[1][ti] = wmma_bf16(a1, b, acc[1][ti]);
    }
  }

  #pragma unroll
  for (int mi = 0; mi < 2; ++mi) {
    #pragma unroll
    for (int ti = 0; ti < 4; ++ti) {
      const int n = n0 + ti * 16 + ln;
      const float bb = bias[n];
      const int h = n >> 6, d = n & 63;
      #pragma unroll
      for (int e = 0; e < 8; ++e) {
        const int token = m0 + mi * 16 + e + 8 * g;   // C layout: row = e + 8g
        const int b_ = token >> 11, s = token & (SEQ - 1);
        const float v = (acc[mi][ti][e] + bb) * scale;
        size_t idx;
        if (mode == 2) idx = ((size_t)(b_ * N_HEADS + h) * D_K + d) * SEQ + s;
        else           idx = ((size_t)(b_ * N_HEADS + h) * SEQ + s) * D_K + d;
        out[idx] = (bf16_t)v;
      }
    }
  }
}

// =====================================================================
// Kernel 2: flash attention, fully in registers (no LDS transpose).
// Scores are computed TRANSPOSED, S^T = K_tile @ Q^T: the C-layout of
// S^T (rows = keys, cols = q) is exactly the A-fragment layout of P for
// the P@V WMMA (element e <-> key 8g+e = S1T[e]; 8+e <-> key 16+8g+e =
// S2T[e]). Softmax stats are per-lane scalars (q = lane&15), combined
// with a single xor-16 shuffle.
// grid = (B*H, S/64), block = 128 (4 waves, one 16-q tile each).
// =====================================================================
__global__ __launch_bounds__(128) void flash_attn(
    const bf16_t* __restrict__ Q, const bf16_t* __restrict__ K,
    const bf16_t* __restrict__ Vt, bf16_t* __restrict__ O) {
  const int lane = threadIdx.x & 31;
  const int wave = threadIdx.x >> 5;
  const int g    = lane >> 4;
  const int ln   = lane & 15;
  const int bh   = blockIdx.x;                 // b*16 + h
  const int q0   = blockIdx.y * 64 + wave * 16;

  const bf16_t* Qh = Q  + (size_t)bh * SEQ * D_K;
  const bf16_t* Kh = K  + (size_t)bh * SEQ * D_K;
  const bf16_t* Vh = Vt + (size_t)bh * D_K * SEQ;

  // Q as the B operand of S^T: column q = ln, K-dim = d.
  const v16bf qb0 = load_b_bf16(Qh + (size_t)(q0 + ln) * D_K,      g);
  const v16bf qb1 = load_b_bf16(Qh + (size_t)(q0 + ln) * D_K + 32, g);

  v8f o[4] = {};
  float m_st = -1e30f, l_st = 0.f;             // stats for q = q0 + ln

  for (int j = 0; j < SEQ; j += 32) {
    // ---- S1T = K[j..j+16) @ Q^T, S2T = K[j+16..j+32) @ Q^T ----
    v8f s1 = {}, s2 = {};
    {
      v16bf ka;
      ka = load_a_bf16(Kh + (size_t)(j + ln) * D_K,           g); s1 = wmma_bf16(ka, qb0, s1);
      ka = load_a_bf16(Kh + (size_t)(j + ln) * D_K + 32,      g); s1 = wmma_bf16(ka, qb1, s1);
      ka = load_a_bf16(Kh + (size_t)(j + 16 + ln) * D_K,      g); s2 = wmma_bf16(ka, qb0, s2);
      ka = load_a_bf16(Kh + (size_t)(j + 16 + ln) * D_K + 32, g); s2 = wmma_bf16(ka, qb1, s2);
    }

    // ---- online softmax over the 32 keys of this step ----
    float t = -1e30f;
    #pragma unroll
    for (int e = 0; e < 8; ++e) t = fmaxf(t, fmaxf(s1[e], s2[e]));
    t = fmaxf(t, __shfl_xor(t, 16, 32));       // combine key halves
    const float mn = fmaxf(m_st, t);
    const float sc = __expf(m_st - mn);
    float rs = 0.f;
    #pragma unroll
    for (int e = 0; e < 8; ++e) {
      s1[e] = __expf(s1[e] - mn);
      s2[e] = __expf(s2[e] - mn);
      rs += s1[e] + s2[e];
    }
    rs += __shfl_xor(rs, 16, 32);
    l_st = l_st * sc + rs;
    m_st = mn;

    // ---- rescale O: O rows are q = e + 8g; fetch that q's scale ----
    float scr[8];
    #pragma unroll
    for (int e = 0; e < 8; ++e) scr[e] = __shfl(sc, e + 8 * g, 32);
    #pragma unroll
    for (int t4 = 0; t4 < 4; ++t4)
      #pragma unroll
      for (int e = 0; e < 8; ++e) o[t4][e] *= scr[e];

    // ---- P A-fragment straight from S^T registers ----
    v16bf pa;
    #pragma unroll
    for (int e = 0; e < 8; ++e) {
      pa[e]     = (bf16_t)s1[e];
      pa[8 + e] = (bf16_t)s2[e];
    }

    // ---- O += P(16x32) @ V(32x64) ----
    #pragma unroll
    for (int t4 = 0; t4 < 4; ++t4) {
      const v16bf vb = load_b_bf16(Vh + (size_t)(t4 * 16 + ln) * SEQ + j, g);
      o[t4] = wmma_bf16(pa, vb, o[t4]);
    }
  }

  // ---- normalize rows and store bf16 [token, h*64+d] ----
  const float inv = 1.f / l_st;
  float invr[8];
  #pragma unroll
  for (int e = 0; e < 8; ++e) invr[e] = __shfl(inv, e + 8 * g, 32);

  const int b_ = bh >> 4, h = bh & 15;
  #pragma unroll
  for (int e = 0; e < 8; ++e) {
    const int s = q0 + e + 8 * g;
    const size_t base = (size_t)(b_ * SEQ + s) * D_MODEL + h * D_K;
    #pragma unroll
    for (int t4 = 0; t4 < 4; ++t4)
      O[base + t4 * 16 + ln] = (bf16_t)(o[t4][e] * invr[e]);
  }
}

// =====================================================================
// Kernel 3: out = LeakyReLU(LayerNorm(attn @ Wo.T + bo + x))
// Block owns 16 full token rows (8 waves x 128 cols). grid = TOKENS/16.
// =====================================================================
__global__ __launch_bounds__(256) void out_proj_ln(
    const bf16_t* __restrict__ A, const bf16_t* __restrict__ Wob,
    const float* __restrict__ bo, const float* __restrict__ X,
    const float* __restrict__ gamma, const float* __restrict__ beta,
    float* __restrict__ out) {
  __shared__ float rsum[16], rsq[16];

  const int lane = threadIdx.x & 31;
  const int wave = threadIdx.x >> 5;
  const int g    = lane >> 4;
  const int ln   = lane & 15;
  const int m0   = blockIdx.x * 16;
  const int n0   = wave * 128;

  if (threadIdx.x < 16) { rsum[threadIdx.x] = 0.f; rsq[threadIdx.x] = 0.f; }
  __syncthreads();

  const bf16_t* arow = A + (size_t)(m0 + ln) * D_MODEL;
  v8f acc[8] = {};

  #pragma unroll 2
  for (int kk = 0; kk < D_MODEL; kk += 32) {
    __builtin_prefetch(arow + kk + 256, 0, 3);
    const v16bf a = load_a_bf16(arow + kk, g);
    #pragma unroll
    for (int t = 0; t < 8; ++t) {
      const int n = n0 + t * 16 + ln;
      const v16bf b = load_b_bf16(Wob + (size_t)n * D_MODEL + kk, g);
      acc[t] = wmma_bf16(a, b, acc[t]);
    }
  }

  // bias + residual (fp32 x)
  #pragma unroll
  for (int t = 0; t < 8; ++t) {
    const int n = n0 + t * 16 + ln;
    const float bb = bo[n];
    #pragma unroll
    for (int e = 0; e < 8; ++e) {
      const int row = m0 + e + 8 * g;
      acc[t][e] += bb + X[(size_t)row * D_MODEL + n];
    }
  }

  // row partial sums over this wave's 128 cols
  float ps[8] = {}, pq[8] = {};
  #pragma unroll
  for (int t = 0; t < 8; ++t)
    #pragma unroll
    for (int e = 0; e < 8; ++e) { const float v = acc[t][e]; ps[e] += v; pq[e] += v * v; }
  #pragma unroll
  for (int e = 0; e < 8; ++e) {
    #pragma unroll
    for (int off = 1; off < 16; off <<= 1) {
      ps[e] += __shfl_xor(ps[e], off, 32);
      pq[e] += __shfl_xor(pq[e], off, 32);
    }
  }
  if (ln == 0) {
    #pragma unroll
    for (int e = 0; e < 8; ++e) {
      atomicAdd(&rsum[e + 8 * g], ps[e]);
      atomicAdd(&rsq[e + 8 * g],  pq[e]);
    }
  }
  __syncthreads();

  // normalize + affine + leaky relu, fp32 store
  #pragma unroll
  for (int t = 0; t < 8; ++t) {
    const int n = n0 + t * 16 + ln;
    const float gm = gamma[n], bt = beta[n];
    #pragma unroll
    for (int e = 0; e < 8; ++e) {
      const int row = e + 8 * g;
      const float mean = rsum[row] * (1.f / D_MODEL);
      const float var  = rsq[row] * (1.f / D_MODEL) - mean * mean;
      const float rstd = rsqrtf(var + 1e-5f);
      float v = (acc[t][e] - mean) * rstd * gm + bt;
      v = (v >= 0.f) ? v : 0.01f * v;
      out[(size_t)(m0 + row) * D_MODEL + n] = v;
    }
  }
}

// =====================================================================
extern "C" void kernel_launch(void* const* d_in, const int* in_sizes, int n_in,
                              void* d_out, int out_size, void* d_ws, size_t ws_size,
                              hipStream_t stream) {
  const float* X     = (const float*)d_in[0];
  const float* Wq    = (const float*)d_in[1];
  const float* bq    = (const float*)d_in[2];
  const float* Wk    = (const float*)d_in[3];
  const float* bk    = (const float*)d_in[4];
  const float* Wv    = (const float*)d_in[5];
  const float* bv    = (const float*)d_in[6];
  const float* Wo    = (const float*)d_in[7];
  const float* bo    = (const float*)d_in[8];
  const float* gamma = (const float*)d_in[9];
  const float* beta  = (const float*)d_in[10];

  const size_t XN = (size_t)TOKENS * D_MODEL;               // 4 Mi
  const size_t WN = (size_t)D_MODEL * D_MODEL;              // 1 Mi
  const size_t QN = (size_t)BATCH * N_HEADS * SEQ * D_K;    // 4 Mi

  bf16_t* Xb  = (bf16_t*)d_ws;    // x in bf16
  bf16_t* Wqb = Xb  + XN;
  bf16_t* Wkb = Wqb + WN;
  bf16_t* Wvb = Wkb + WN;
  bf16_t* Wob = Wvb + WN;
  bf16_t* Qb  = Wob + WN;         // [B,H,S,D], pre-scaled by 1/sqrt(D_K)
  bf16_t* Kb  = Qb  + QN;         // [B,H,S,D]
  bf16_t* Vb  = Kb  + QN;         // [B,H,D,S] (transposed)
  bf16_t* Ab  = Vb  + QN;         // attn out, [token, 1024]

  // ---- bulk fp32 -> bf16 ----
  cvt_bf16<<<dim3(XN / (8 * 256)), 256, 0, stream>>>(X,  Xb,  (int)XN);
  cvt_bf16<<<dim3(WN / (8 * 256)), 256, 0, stream>>>(Wq, Wqb, (int)WN);
  cvt_bf16<<<dim3(WN / (8 * 256)), 256, 0, stream>>>(Wk, Wkb, (int)WN);
  cvt_bf16<<<dim3(WN / (8 * 256)), 256, 0, stream>>>(Wv, Wvb, (int)WN);
  cvt_bf16<<<dim3(WN / (8 * 256)), 256, 0, stream>>>(Wo, Wob, (int)WN);

  // ---- QKV projections (block tile 128x128) ----
  dim3 gGemm(TOKENS / 128, D_MODEL / 128);
  qkv_gemm<<<gGemm, 256, 0, stream>>>(Xb, Wqb, bq, Qb, 0, 0.125f);
  qkv_gemm<<<gGemm, 256, 0, stream>>>(Xb, Wkb, bk, Kb, 1, 1.0f);
  qkv_gemm<<<gGemm, 256, 0, stream>>>(Xb, Wvb, bv, Vb, 2, 1.0f);

  // ---- attention ----
  flash_attn<<<dim3(BATCH * N_HEADS, SEQ / 64), 128, 0, stream>>>(Qb, Kb, Vb, Ab);

  // ---- output projection + residual + LayerNorm + LeakyReLU ----
  out_proj_ln<<<dim3(TOKENS / 16), 256, 0, stream>>>(Ab, Wob, bo, X, gamma, beta,
                                                     (float*)d_out);
}
